// VSAMemory_38792144617755
// MI455X (gfx1250) — matrix operand order repaired and checked by reading, not compiled, not run on hardware
//
#include <hip/hip_runtime.h>
#include <cstdint>
#include <cstddef>

// ---------------------------------------------------------------------------
// VSA holographic memory for MI455X (gfx1250, wave32, WMMA)
// Heavy op: two (1024 x 65536) similarity GEMMs (K=256) -> top-16 per row,
// done with V_WMMA_F32_16X16X32_BF16, B tiles streamed with
// GLOBAL_LOAD_ASYNC_TO_LDS_B128 (double-buffered, ASYNCcnt-tracked), and the
// K-loop software-pipelined so each WMMA overlaps the next fragment's LDS
// loads (partial DScnt waits instead of full drains).
// ---------------------------------------------------------------------------

typedef __attribute__((ext_vector_type(16))) __bf16 v16bf;
typedef __attribute__((ext_vector_type(8)))  float  v8f;

#define B_ROWS   1024
#define DIM      256
#define NSLOTS   65536
#define KTOP     16
#define DECAYF   0.995f
#define EPSF     1e-8f

#define NSPLIT     16
#define SLICE      (NSLOTS / NSPLIT)   // 4096 slots per block
#define TILE_COLS  64                  // slots per inner iteration (4 waves x 16)
#define ITERS      (SLICE / TILE_COLS) // 64
#define TSIMS      128                 // 4 wave32
#define TILE_U4    (TILE_COLS * DIM / 8)  // 2048 x uint4 = 32 KB per tile

// -------- CDNA5 async copy helpers (ASYNCcnt path, CDNA5 ISA) -----
__device__ __forceinline__ void async_copy_b128(unsigned lds_off, const void* gptr) {
  // per-lane: LDS[lds_off] = MEM[gptr], 16 bytes, tracked by ASYNCcnt
  asm volatile("global_load_async_to_lds_b128 %0, %1, off"
               :: "v"(lds_off), "v"(gptr)
               : "memory");
}
__device__ __forceinline__ void wait_async0() {
  asm volatile("s_wait_asynccnt 0x0" ::: "memory");
}
__device__ __forceinline__ unsigned lds_addr(const void* p) {
  // generic LDS pointer: low 32 bits are the wave-relative LDS byte offset
  return (unsigned)(uintptr_t)p;
}

// ---------------------------------------------------------------------------
// addresses (fp32, pre-normalized in setup) -> bf16 staging copy
// ---------------------------------------------------------------------------
__global__ void convert_addr_bf16(const float* __restrict__ a,
                                  __bf16* __restrict__ o, int n) {
  int i = blockIdx.x * blockDim.x + threadIdx.x;
  int s = gridDim.x * blockDim.x;
  for (; i < n; i += s) o[i] = (__bf16)a[i];
}

// ---------------------------------------------------------------------------
// normalize key/value/query rows; emit q_n (f32), key/query bf16 copies for
// the WMMA search, and bound = circular_conv(key_n, val_n) (time domain,
// exact equivalent of rfft-multiply-irfft for real signals).
// ---------------------------------------------------------------------------
__global__ void norm_bind_kernel(const float* __restrict__ key,
                                 const float* __restrict__ val,
                                 const float* __restrict__ qry,
                                 float* __restrict__ q_n,
                                 float* __restrict__ bound,
                                 __bf16* __restrict__ keyb,
                                 __bf16* __restrict__ qb) {
  __shared__ float kn[DIM], vn[DIM], red[DIM];
  const int b = blockIdx.x, t = threadIdx.x;

  // key
  float kx = key[b * DIM + t];
  red[t] = kx * kx; __syncthreads();
  for (int o = DIM / 2; o > 0; o >>= 1) { if (t < o) red[t] += red[t + o]; __syncthreads(); }
  float knv = kx / (sqrtf(red[0]) + EPSF);
  __syncthreads();
  kn[t] = knv; keyb[b * DIM + t] = (__bf16)knv;

  // value
  float vx = val[b * DIM + t];
  red[t] = vx * vx; __syncthreads();
  for (int o = DIM / 2; o > 0; o >>= 1) { if (t < o) red[t] += red[t + o]; __syncthreads(); }
  float vnv = vx / (sqrtf(red[0]) + EPSF);
  __syncthreads();
  vn[t] = vnv;

  // query
  float qx = qry[b * DIM + t];
  red[t] = qx * qx; __syncthreads();
  for (int o = DIM / 2; o > 0; o >>= 1) { if (t < o) red[t] += red[t + o]; __syncthreads(); }
  float qnv = qx / (sqrtf(red[0]) + EPSF);
  q_n[b * DIM + t] = qnv; qb[b * DIM + t] = (__bf16)qnv;
  __syncthreads();

  // circular convolution: bound[i] = sum_j kn[j] * vn[(i-j) mod 256]
  float acc = 0.f;
  #pragma unroll 8
  for (int j = 0; j < DIM; ++j) acc += kn[j] * vn[(t - j) & (DIM - 1)];
  bound[b * DIM + t] = acc;
}

// ---------------------------------------------------------------------------
// new_mem = memory * DECAY (float4 streaming)
// ---------------------------------------------------------------------------
__global__ void decay_kernel(const float4* __restrict__ m,
                             float4* __restrict__ o, int n4) {
  int i = blockIdx.x * blockDim.x + threadIdx.x;
  int s = gridDim.x * blockDim.x;
  for (; i < n4; i += s) {
    float4 v = m[i];
    v.x *= DECAYF; v.y *= DECAYF; v.z *= DECAYF; v.w *= DECAYF;
    o[i] = v;
  }
}

// ---------------------------------------------------------------------------
// Similarity GEMM + running per-row top-16.
// Block: 16 batch rows x SLICE slots, 4 waves. A fragments persist in
// registers; B tiles double-buffered in LDS via async copies. Fragment
// packing (contiguous K, same order as B): A lanes 0-15 hold K 0..15, lanes 16-31
// K 16..31; B lanes 0-15 K=e, lanes 16-31 K=16+e.
// ---------------------------------------------------------------------------
__global__ __launch_bounds__(TSIMS)
void sims_topk_kernel(const __bf16* __restrict__ Q,   // (B_ROWS, DIM)
                      const __bf16* __restrict__ A,   // (NSLOTS, DIM)
                      float* __restrict__ pval,       // (B_ROWS, NSPLIT, KTOP)
                      int*   __restrict__ pidx) {
  __shared__ __align__(16) __bf16 alds[16][DIM];           //  8 KB
  __shared__ __align__(16) __bf16 blds[2][TILE_COLS][DIM]; // 64 KB (double buffer)
  __shared__ float slds[16][TILE_COLS + 1];                //  4 KB (padded)
  __shared__ float topv[16][KTOP + 1];
  __shared__ int   topi[16][KTOP + 1];

  const int tid  = threadIdx.x;
  const int lane = tid & 31;
  const int wave = tid >> 5;
  const int rBase = blockIdx.x * 16;
  const int sBase = blockIdx.y * SLICE;

  const uint4* gsrc = (const uint4*)(A + (size_t)sBase * DIM);
  const unsigned ldsB[2] = { lds_addr(&blds[0][0][0]), lds_addr(&blds[1][0][0]) };

  // kick async copy of tile 0 into buffer 0 (overlaps with A staging below)
  {
    const uint4* s = gsrc;
    #pragma unroll
    for (int i = 0; i < TILE_U4 / TSIMS; ++i) {
      const int idx = tid + i * TSIMS;
      async_copy_b128(ldsB[0] + idx * 16, s + idx);
    }
  }

  // stage the 16 query rows into LDS (once)
  {
    const uint4* src = (const uint4*)(Q + (size_t)rBase * DIM);
    uint4* dst = (uint4*)&alds[0][0];
    for (int i = tid; i < 16 * DIM / 8; i += TSIMS) dst[i] = src[i];
  }
  if (tid < 16) {
    #pragma unroll
    for (int j = 0; j < KTOP; ++j) { topv[tid][j] = -1e30f; topi[tid][j] = 0x7fffffff; }
  }
  __syncthreads();

  union Frag { v16bf v; uint4 u[2]; };
  Frag afrag[8];
  {
    const int m  = lane & 15;
    const int o0 = (lane < 16) ? 0 : 16;  // K sub-chunk base for this half-wave
    #pragma unroll
    for (int kb = 0; kb < 8; ++kb) {
      const int kBase = kb * 32;
      afrag[kb].u[0] = *(const uint4*)&alds[m][kBase + o0];
      afrag[kb].u[1] = *(const uint4*)&alds[m][kBase + o0 + 8];
    }
  }

  const int n    = lane & 15;
  const int col  = wave * 16 + n;
  const int ko   = (lane < 16) ? 0 : 16;   // B layout K base for this half-wave
  const int mBase = (lane < 16) ? 0 : 8;   // C layout row base

  for (int it = 0; it < ITERS; ++it) {
    const int cur = it & 1;
    wait_async0();        // this wave's async stores to LDS for tile `it` done
    __syncthreads();      // all waves done -> tile `it` visible, prev scan done

    if (it + 1 < ITERS) { // issue tile it+1 into the other buffer (overlapped)
      const uint4* s = gsrc + (size_t)(it + 1) * TILE_U4;
      #pragma unroll
      for (int i = 0; i < TILE_U4 / TSIMS; ++i) {
        const int idx = tid + i * TSIMS;
        async_copy_b128(ldsB[cur ^ 1] + idx * 16, s + idx);
      }
      if (it + 2 < ITERS) { // warm L2 for tile it+2 (global_prefetch_b8)
        const char* nx = (const char*)(gsrc + (size_t)(it + 2) * TILE_U4);
        __builtin_prefetch(nx + tid * 128, 0, 3);
        __builtin_prefetch(nx + 16384 + tid * 128, 0, 3);
      }
    }

    // 16(batch) x 16(slot) tile for this wave. Software-pipelined K-loop:
    // fragment kb+1's ds_load_b128 pair is issued BEFORE wmma kb, so the
    // wait before each wmma only drains to DScnt<=2 and LDS latency hides
    // behind the previous WMMA.
    {
      Frag bc;
      bc.u[0] = *(const uint4*)&blds[cur][col][ko];
      bc.u[1] = *(const uint4*)&blds[cur][col][ko + 8];
      v8f c = {0.f, 0.f, 0.f, 0.f, 0.f, 0.f, 0.f, 0.f};
      #pragma unroll
      for (int kb = 0; kb < 8; ++kb) {
        Frag bn;
        if (kb < 7) {
          const int kBase = (kb + 1) * 32 + ko;
          bn.u[0] = *(const uint4*)&blds[cur][col][kBase];
          bn.u[1] = *(const uint4*)&blds[cur][col][kBase + 8];
        }
        c = __builtin_amdgcn_wmma_f32_16x16x32_bf16(
                false, afrag[kb].v, false, bc.v, (short)0, c, false, false);
        if (kb < 7) bc = bn;
      }
      #pragma unroll
      for (int r = 0; r < 8; ++r) slds[mBase + r][wave * 16 + n] = c[r];
    }
    __syncthreads();  // slds ready

    // running top-16 per row (sorted-descending insertion lists in LDS)
    if (tid < 16) {
      const int row = tid;
      float vmin = topv[row][KTOP - 1];
      const int gbase = sBase + it * TILE_COLS;
      for (int cx = 0; cx < TILE_COLS; ++cx) {
        const float v = slds[row][cx];
        if (v > vmin) {
          const int gi = gbase + cx;
          int p = KTOP - 1;
          while (p > 0 && topv[row][p - 1] < v) {
            topv[row][p] = topv[row][p - 1];
            topi[row][p] = topi[row][p - 1];
            --p;
          }
          topv[row][p] = v; topi[row][p] = gi;
          vmin = topv[row][KTOP - 1];
        }
      }
    }
  }
  __syncthreads();
  if (tid < 16) {
    const size_t base = ((size_t)(rBase + tid) * NSPLIT + blockIdx.y) * KTOP;
    #pragma unroll
    for (int j = 0; j < KTOP; ++j) { pval[base + j] = topv[tid][j]; pidx[base + j] = topi[tid][j]; }
  }
}

// ---------------------------------------------------------------------------
// Merge NSPLIT partial top-16 lists per row -> final 16 indices (descending,
// ties -> lower slot index, matching lax.top_k). One wave32 per row.
// ---------------------------------------------------------------------------
__global__ void merge_topk_kernel(const float* __restrict__ pval,
                                  const int* __restrict__ pidx,
                                  int* __restrict__ outidx) {
  const int b = blockIdx.x;
  const int lane = threadIdx.x;        // blockDim = 32
  float lv[8]; int li[8];
  const size_t base = (size_t)b * (NSPLIT * KTOP);
  #pragma unroll
  for (int j = 0; j < 8; ++j) { const int c = lane * 8 + j; lv[j] = pval[base + c]; li[j] = pidx[base + c]; }

  for (int sel = 0; sel < KTOP; ++sel) {
    float bv = -1e38f; int bi = 0x7fffffff; int bslot = 0;
    #pragma unroll
    for (int j = 0; j < 8; ++j)
      if (lv[j] > bv || (lv[j] == bv && li[j] < bi)) { bv = lv[j]; bi = li[j]; bslot = j; }
    int blane = lane;
    #pragma unroll
    for (int off = 16; off > 0; off >>= 1) {
      const float ov = __shfl_xor(bv, off, 32);
      const int   oi = __shfl_xor(bi, off, 32);
      const int   ol = __shfl_xor(blane, off, 32);
      const int   os = __shfl_xor(bslot, off, 32);
      if (ov > bv || (ov == bv && oi < bi)) { bv = ov; bi = oi; blane = ol; bslot = os; }
    }
    if (lane == 0) outidx[b * KTOP + sel] = bi;
    #pragma unroll
    for (int j = 0; j < 8; ++j)
      if (lane == blane && bslot == j) lv[j] = -1e38f;   // retire winner
  }
}

// ---------------------------------------------------------------------------
// scatter-add bound rows into decayed memory (duplicates accumulate, as in
// the reference's .at[idx].add). global_atomic_add_f32.
// ---------------------------------------------------------------------------
__global__ void scatter_kernel(float* __restrict__ nmem,
                               const int* __restrict__ idxw,
                               const float* __restrict__ bound) {
  const int b = blockIdx.x, t = threadIdx.x;
  const float v = bound[b * DIM + t];
  #pragma unroll
  for (int k = 0; k < KTOP; ++k) {
    const int s = idxw[b * KTOP + k];
    atomicAdd(&nmem[(size_t)s * DIM + t], v);
  }
}

// ---------------------------------------------------------------------------
// read: gather-sum 16 rows, hrr_unbind via exact 256-point DFT (twiddle
// table; angles are multiples of 2*pi/256 so lookups are exact), normalize.
// ---------------------------------------------------------------------------
__global__ void read_unbind_kernel(const float* __restrict__ nmem,
                                   const int* __restrict__ idxr,
                                   const float* __restrict__ q_n,
                                   float* __restrict__ out) {
  __shared__ float qs[DIM], cs[DIM], ct[DIM], st[DIM];
  __shared__ float Bre[DIM / 2 + 1], Bim[DIM / 2 + 1];
  __shared__ float red[DIM];
  const int b = blockIdx.x, t = threadIdx.x;

  float acc = 0.f;
  #pragma unroll
  for (int k = 0; k < KTOP; ++k) {
    const int s = idxr[b * KTOP + k];
    acc += nmem[(size_t)s * DIM + t];
  }
  cs[t] = acc;
  qs[t] = q_n[b * DIM + t];
  const float ang = (2.0f * 3.14159265358979323846f / DIM) * (float)t;
  ct[t] = cosf(ang); st[t] = sinf(ang);
  __syncthreads();

  if (t <= DIM / 2) {   // rfft of q and content; Wiener division
    float are = 0.f, aim = 0.f, cre = 0.f, cim = 0.f;
    for (int x = 0; x < DIM; ++x) {
      const int id = (t * x) & (DIM - 1);
      const float co = ct[id], si = st[id];
      const float qv = qs[x], cv = cs[x];
      are += qv * co; aim -= qv * si;
      cre += cv * co; cim -= cv * si;
    }
    const float den = are * are + aim * aim + EPSF;
    Bre[t] = (cre * are + cim * aim) / den;
    Bim[t] = (cim * are - cre * aim) / den;
  }
  __syncthreads();

  // irfft(B_hat, 256)
  float x = Bre[0] + ((t & 1) ? -Bre[DIM / 2] : Bre[DIM / 2]);
  for (int f = 1; f < DIM / 2; ++f) {
    const int id = (f * t) & (DIM - 1);
    x += 2.0f * (Bre[f] * ct[id] - Bim[f] * st[id]);
  }
  x *= (1.0f / DIM);

  red[t] = x * x; __syncthreads();
  for (int o = DIM / 2; o > 0; o >>= 1) { if (t < o) red[t] += red[t + o]; __syncthreads(); }
  out[b * DIM + t] = x / (sqrtf(red[0]) + EPSF);
}

// ---------------------------------------------------------------------------
extern "C" void kernel_launch(void* const* d_in, const int* in_sizes, int n_in,
                              void* d_out, int out_size, void* d_ws, size_t ws_size,
                              hipStream_t stream) {
  (void)in_sizes; (void)n_in; (void)out_size; (void)ws_size;

  const float* key       = (const float*)d_in[0];
  const float* value     = (const float*)d_in[1];
  const float* query     = (const float*)d_in[2];
  const float* addresses = (const float*)d_in[3];
  const float* memory    = (const float*)d_in[4];
  float* out = (float*)d_out;

  // workspace carve-out (256B aligned regions)
  char* ws = (char*)d_ws;
  size_t off = 0;
  auto carve = [&](size_t bytes) -> char* {
    char* p = ws + off;
    off += (bytes + 255) & ~(size_t)255;
    return p;
  };
  __bf16* addrb = (__bf16*)carve((size_t)NSLOTS * DIM * sizeof(__bf16));   // 32 MB
  float*  nmem  = (float*) carve((size_t)NSLOTS * DIM * sizeof(float));    // 64 MB
  float*  q_n   = (float*) carve((size_t)B_ROWS * DIM * sizeof(float));
  float*  bound = (float*) carve((size_t)B_ROWS * DIM * sizeof(float));
  __bf16* keyb  = (__bf16*)carve((size_t)B_ROWS * DIM * sizeof(__bf16));
  __bf16* qb    = (__bf16*)carve((size_t)B_ROWS * DIM * sizeof(__bf16));
  float*  pval  = (float*) carve((size_t)B_ROWS * NSPLIT * KTOP * sizeof(float));
  int*    pidx  = (int*)   carve((size_t)B_ROWS * NSPLIT * KTOP * sizeof(int));
  int*    idxw  = (int*)   carve((size_t)B_ROWS * KTOP * sizeof(int));
  int*    idxr  = (int*)   carve((size_t)B_ROWS * KTOP * sizeof(int));

  // 1. stage addresses as bf16
  convert_addr_bf16<<<2048, 256, 0, stream>>>(addresses, addrb, NSLOTS * DIM);
  // 2. normalize + bind
  norm_bind_kernel<<<B_ROWS, DIM, 0, stream>>>(key, value, query, q_n, bound, keyb, qb);
  // 3. decayed memory copy
  decay_kernel<<<4096, 256, 0, stream>>>((const float4*)memory, (float4*)nmem,
                                         NSLOTS * DIM / 4);
  // 4. write-path similarity search + top-16
  dim3 g(B_ROWS / 16, NSPLIT);
  sims_topk_kernel<<<g, TSIMS, 0, stream>>>(keyb, addrb, pval, pidx);
  merge_topk_kernel<<<B_ROWS, 32, 0, stream>>>(pval, pidx, idxw);
  // 5. read-path similarity search + top-16 (reuses partial buffers; stream-ordered)
  sims_topk_kernel<<<g, TSIMS, 0, stream>>>(qb, addrb, pval, pidx);
  merge_topk_kernel<<<B_ROWS, 32, 0, stream>>>(pval, pidx, idxr);
  // 6. scatter-add writes into decayed memory
  scatter_kernel<<<B_ROWS, DIM, 0, stream>>>(nmem, idxw, bound);
  // 7. gather + unbind + normalize
  read_unbind_kernel<<<B_ROWS, DIM, 0, stream>>>(nmem, idxr, q_n, out);
}
